// GINStack_50989851738507
// MI455X (gfx1250) — matrix operand order repaired and checked by reading, not compiled, not run on hardware
//
#include <hip/hip_runtime.h>

typedef __attribute__((ext_vector_type(16))) _Float16 v16h;
typedef __attribute__((ext_vector_type(8)))  float    v8f;
typedef __attribute__((ext_vector_type(4)))  unsigned v4u;
typedef __attribute__((ext_vector_type(8)))  unsigned v8u;

#define DN    128     // node feature dim
#define DE    64      // edge feature dim
#define KTF   192     // DN + DE
#define HIDD  128
#define OUTD  128
#define LNUM  3
#define NCAT  384     // HIDD * LNUM

#define TILE_M 128
#define TILE_N 128
#define KCHUNK 64
#define LDS_LD 72     // 64 data + 8 pad halves; 144B row stride = 9*16B (16B aligned)

// Single shared struct per kernel so LDS offsets are well-defined from base 0.
struct alignas(16) TileLds {
  _Float16 A[TILE_M * LDS_LD];
  _Float16 B[TILE_N * LDS_LD];
};
#define LDS_B_OFF ((unsigned)(TILE_M * LDS_LD * 2))   // byte offset of B region

// ---------------------------------------------------------------------------
// Tensor Data Mover: async DMA of a 2D f16 tile [128 rows x 64 halves] from
// global (row stride = row_stride_elems halves) into LDS at lds_byte_off with
// LDS padding of 4 DWORDs after every 32 DWORDs (=> 72-half row stride).
// D# layout per CDNA5 ISA ch.8 (group0 128b, group1 256b; groups 2/3 = NULL).
// Issued by one wave; EXEC ignored by TDM; tracked via TENSORcnt.
// ---------------------------------------------------------------------------
static __device__ __forceinline__ void tdm_load_b(
    unsigned lds_byte_off, const _Float16* gptr, unsigned row_stride_elems) {
  unsigned long long ga = (unsigned long long)(uintptr_t)gptr;
  v4u g0;
  g0[0] = 1u;                                      // count=1 (valid), user mode
  g0[1] = lds_byte_off;                            // lds_addr
  g0[2] = (unsigned)ga;                            // global_addr[31:0]
  g0[3] = ((unsigned)(ga >> 32) & 0x1FFFFFFu)      // global_addr[56:32]
          | (2u << 30);                            // type = 2 ("image")
  v8u g1;
  g1[0] = (1u << 16)                               // data_size = 2 bytes
        | (1u << 20)                               // pad_enable
        | (4u << 22)                               // pad_interval: 32 DWORDs
        | (3u << 25);                              // pad_amount:   4 DWORDs
  g1[1] = (64u & 0xFFFFu) << 16;                   // tensor_dim0[15:0] = 64
  g1[2] = (128u & 0xFFFFu) << 16;                  // dim0[31:16]=0, tensor_dim1[15:0]=128
  g1[3] = (64u << 16);                             // dim1[31:16]=0, tile_dim0 = 64
  g1[4] = 128u;                                    // tile_dim1 = 128, tile_dim2 = 0
  g1[5] = row_stride_elems;                        // tensor_dim0_stride[31:0]
  g1[6] = 0u;                                      // stride[47:32]=0, dim1_stride lo=0
  g1[7] = 0u;
  asm volatile("tensor_load_to_lds %0, %1" :: "s"(g0), "s"(g1) : "memory");
}

// ---------------------------------------------------------------------------
// WMMA fragment loads from LDS (f16, 16x16x32, wave32 layouts per ISA 7.12.2).
// Both fragments are two contiguous 16B chunks => explicit uint4 loads force
// ds_load_b128 codegen.
// ---------------------------------------------------------------------------
static __device__ __forceinline__ v16h load_a_frag(const _Float16* As, int m0, int k0, int lane) {
  int m  = m0 + (lane & 15);
  int hi = (lane >> 4) << 3;               // 0 or 8
  union { v16h h; uint4 q[2]; } f;
  const _Float16* base = As + m * LDS_LD + k0 + hi;
  f.q[0] = *(const uint4*)(base);          // K = k0+hi   .. +7
  f.q[1] = *(const uint4*)(base + 16);     // K = k0+16+hi .. +7
  return f.h;
}

static __device__ __forceinline__ v16h load_b_frag(const _Float16* Bs, int k0, int n0, int lane) {
  int n  = n0 + (lane & 15);
  int hi = (lane >> 4) << 3;               // 0 or 8
  union { v16h h; uint4 q[2]; } f;
  const _Float16* base = Bs + n * LDS_LD + k0 + hi;
  f.q[0] = *(const uint4*)(base);          // K = k0+hi .. +7
  f.q[1] = *(const uint4*)(base + 16);     // K = k0+16+hi .. +7
  return f.h;
}

// One k-step: preload all 8 B fragments (16 ds_load_b128), then chain 8 WMMAs.
static __device__ __forceinline__ void mma_step(const _Float16* As, const _Float16* Bs,
                                                int m0, int k0, int lane, v8f acc[8]) {
  v16h a = load_a_frag(As, m0, k0, lane);
  v16h b[8];
#pragma unroll
  for (int nt = 0; nt < 8; ++nt) b[nt] = load_b_frag(Bs, k0, nt * 16, lane);
#pragma unroll
  for (int nt = 0; nt < 8; ++nt)
    acc[nt] = __builtin_amdgcn_wmma_f32_16x16x32_f16(
        false, a, false, b[nt], (short)0, acc[nt], false, false);
}

static __device__ __forceinline__ void atomAddF(float* p, float v) {
  unsafeAtomicAdd(p, v);   // global_atomic_add_f32 (no-return) on gfx1250
}

// ---------------------------------------------------------------------------
// Utility kernels
// ---------------------------------------------------------------------------
__global__ void k_zero(float* __restrict__ p, size_t n) {
  size_t i  = (size_t)blockIdx.x * blockDim.x + threadIdx.x;
  size_t st = (size_t)gridDim.x * blockDim.x;
  for (; i < n; i += st) p[i] = 0.f;
}

__global__ void k_f2h(const float* __restrict__ s, _Float16* __restrict__ d, int n) {
  int i  = blockIdx.x * blockDim.x + threadIdx.x;
  int st = gridDim.x * blockDim.x;
  for (; i < n; i += st) d[i] = (_Float16)s[i];
}

__global__ void k_copy_i32(const int* __restrict__ s, int* __restrict__ d, int n) {
  int i = blockIdx.x * blockDim.x + threadIdx.x;
  if (i < n) d[i] = s[i];
}
__global__ void k_widen_i64(const int* __restrict__ s, long long* __restrict__ d, int n) {
  int i = blockIdx.x * blockDim.x + threadIdx.x;
  if (i < n) d[i] = (long long)s[i];
}

// ---------------------------------------------------------------------------
// Edge GEMM + scatter:  agg[src[e], col0+n] += relu( tf[e,:] . Wcat[:, col0+n] + b )
// tf = concat(tgt[e,0:128], edge[e,0:64]).  grid.x tiles edges, grid.y = layer.
// ---------------------------------------------------------------------------
__global__ __launch_bounds__(256) void k_edge_gemm_scatter(
    const float* __restrict__ tgt, const float* __restrict__ edgef,
    const _Float16* __restrict__ WcatT,   // [384][192] halves ([n][k])
    const float* __restrict__ bcat,       // [384]
    const int* __restrict__ srcidx,
    float* __restrict__ agg, int E) {
  __shared__ TileLds lt;

  const int tid  = threadIdx.x;
  const int lane = tid & 31;
  const int wave = tid >> 5;
  const int e0   = blockIdx.x * TILE_M;
  const int col0 = blockIdx.y * TILE_N;   // layer * HIDD
  const int m0   = wave * 16;

  v8f acc[8] = {};

  for (int kc = 0; kc < KTF / KCHUNK; ++kc) {
    const int kbase = kc * KCHUNK;
    __syncthreads();                      // previous chunk fully consumed
    if (wave == 0)                        // kick B-tile DMA (overlaps A staging)
      tdm_load_b(LDS_B_OFF, WcatT + (size_t)col0 * KTF + kbase, KTF);
    // stage A: 128 edges x 64 K, fp32 -> f16
    for (int idx = tid; idx < TILE_M * KCHUNK; idx += 256) {
      int m = idx >> 6, kl = idx & 63;
      int kg = kbase + kl;
      int e  = e0 + m;
      float v = 0.f;
      if (e < E) v = (kg < DN) ? tgt[(size_t)e * DN + kg]
                               : edgef[(size_t)e * DE + (kg - DN)];
      lt.A[m * LDS_LD + kl] = (_Float16)v;
    }
    if (wave == 0) __builtin_amdgcn_s_wait_tensorcnt(0);
    __syncthreads();
#pragma unroll
    for (int ks = 0; ks < 2; ++ks)
      mma_step(lt.A, lt.B, m0, ks * 32, lane, acc);
  }

  // epilogue: bias + relu + scatter atomic add
  const int hi = lane >> 4, nl = lane & 15;
#pragma unroll
  for (int nt = 0; nt < 8; ++nt) {
    int n_g = col0 + nt * 16 + nl;
    float bia = bcat[n_g];
#pragma unroll
    for (int r = 0; r < 8; ++r) {
      int e = e0 + m0 + r + 8 * hi;
      if (e < E) {
        float v = acc[nt][r] + bia;
        v = v > 0.f ? v : 0.f;
        atomAddF(&agg[(size_t)srcidx[e] * NCAT + n_g], v);
      }
    }
  }
}

// ---------------------------------------------------------------------------
// z = (hprev + agg_l) @ W1^T, fused per-column sum / sumsq for batchnorm.
// ---------------------------------------------------------------------------
__global__ __launch_bounds__(256) void k_node_gemm_bnstat(
    const float* __restrict__ hprev, int ldh,
    const float* __restrict__ aggl,          // stride NCAT
    const _Float16* __restrict__ W1t,        // [128][128] halves ([n][k])
    float* __restrict__ z,
    float* __restrict__ bnsum, float* __restrict__ bnsq, int Nn) {
  __shared__ TileLds lt;

  const int tid = threadIdx.x, lane = tid & 31, wave = tid >> 5;
  const int r0  = blockIdx.x * TILE_M;
  const int m0  = wave * 16;

  v8f acc[8] = {};

  for (int kc = 0; kc < HIDD / KCHUNK; ++kc) {
    const int kbase = kc * KCHUNK;
    __syncthreads();
    if (wave == 0) tdm_load_b(LDS_B_OFF, W1t + kbase, HIDD);
    for (int idx = tid; idx < TILE_M * KCHUNK; idx += 256) {
      int m = idx >> 6, kl = idx & 63;
      int kg = kbase + kl;
      int row = r0 + m;
      float v = 0.f;
      if (row < Nn) v = hprev[(size_t)row * ldh + kg] + aggl[(size_t)row * NCAT + kg];
      lt.A[m * LDS_LD + kl] = (_Float16)v;
    }
    if (wave == 0) __builtin_amdgcn_s_wait_tensorcnt(0);
    __syncthreads();
#pragma unroll
    for (int ks = 0; ks < 2; ++ks)
      mma_step(lt.A, lt.B, m0, ks * 32, lane, acc);
  }

  const int hi = lane >> 4, nl = lane & 15;
#pragma unroll
  for (int nt = 0; nt < 8; ++nt) {
    int n_g = nt * 16 + nl;
    float sl = 0.f, sq = 0.f;
#pragma unroll
    for (int r = 0; r < 8; ++r) {
      int row = r0 + m0 + r + 8 * hi;
      if (row < Nn) {
        float v = acc[nt][r];
        z[(size_t)row * HIDD + n_g] = v;
        sl += v; sq += v * v;
      }
    }
    atomAddF(&bnsum[n_g], sl);
    atomAddF(&bnsq[n_g], sq);
  }
}

__global__ void k_bn_finalize(const float* __restrict__ bnsum, const float* __restrict__ bnsq,
                              const float* __restrict__ g, const float* __restrict__ b,
                              float* __restrict__ bnA, float* __restrict__ bnB, int Nn) {
  int i = threadIdx.x;
  if (i < HIDD) {
    float inv = 1.f / (float)Nn;
    float mu  = bnsum[i] * inv;
    float var = bnsq[i] * inv - mu * mu;
    float a   = g[i] * rsqrtf(var + 1e-5f);
    bnA[i] = a;
    bnB[i] = b[i] - mu * a;
  }
}

// ---------------------------------------------------------------------------
// h = relu(z * bnA + bnB) @ W2^T  -> concat slice (stride NCAT)
// ---------------------------------------------------------------------------
__global__ __launch_bounds__(256) void k_node_gemm_out(
    const float* __restrict__ z,
    const float* __restrict__ bnA, const float* __restrict__ bnB,
    const _Float16* __restrict__ W2t,   // [128][128] halves
    float* __restrict__ outp, int Nn) {
  __shared__ TileLds lt;

  const int tid = threadIdx.x, lane = tid & 31, wave = tid >> 5;
  const int r0  = blockIdx.x * TILE_M;
  const int m0  = wave * 16;

  v8f acc[8] = {};

  for (int kc = 0; kc < HIDD / KCHUNK; ++kc) {
    const int kbase = kc * KCHUNK;
    __syncthreads();
    if (wave == 0) tdm_load_b(LDS_B_OFF, W2t + kbase, HIDD);
    for (int idx = tid; idx < TILE_M * KCHUNK; idx += 256) {
      int m = idx >> 6, kl = idx & 63;
      int kg = kbase + kl;
      int row = r0 + m;
      float v = 0.f;
      if (row < Nn) {
        v = z[(size_t)row * HIDD + kg] * bnA[kg] + bnB[kg];
        v = v > 0.f ? v : 0.f;
      }
      lt.A[m * LDS_LD + kl] = (_Float16)v;
    }
    if (wave == 0) __builtin_amdgcn_s_wait_tensorcnt(0);
    __syncthreads();
#pragma unroll
    for (int ks = 0; ks < 2; ++ks)
      mma_step(lt.A, lt.B, m0, ks * 32, lane, acc);
  }

  const int hi = lane >> 4, nl = lane & 15;
#pragma unroll
  for (int nt = 0; nt < 8; ++nt) {
    int n_g = nt * 16 + nl;
#pragma unroll
    for (int r = 0; r < 8; ++r) {
      int row = r0 + m0 + r + 8 * hi;
      if (row < Nn) outp[(size_t)row * NCAT + n_g] = acc[nt][r];
    }
  }
}

// ---------------------------------------------------------------------------
// out = concat[N,384] @ Wout^T + bout
// ---------------------------------------------------------------------------
__global__ __launch_bounds__(256) void k_final_gemm(
    const float* __restrict__ concat,
    const _Float16* __restrict__ WoutT,   // [128][384] halves ([n][k])
    const float* __restrict__ bout,
    float* __restrict__ out, int Nn) {
  __shared__ TileLds lt;

  const int tid = threadIdx.x, lane = tid & 31, wave = tid >> 5;
  const int r0  = blockIdx.x * TILE_M;
  const int m0  = wave * 16;

  v8f acc[8] = {};

  for (int kc = 0; kc < NCAT / KCHUNK; ++kc) {
    const int kbase = kc * KCHUNK;
    __syncthreads();
    if (wave == 0) tdm_load_b(LDS_B_OFF, WoutT + kbase, NCAT);
    for (int idx = tid; idx < TILE_M * KCHUNK; idx += 256) {
      int m = idx >> 6, kl = idx & 63;
      int kg = kbase + kl;
      int row = r0 + m;
      float v = (row < Nn) ? concat[(size_t)row * NCAT + kg] : 0.f;
      lt.A[m * LDS_LD + kl] = (_Float16)v;
    }
    if (wave == 0) __builtin_amdgcn_s_wait_tensorcnt(0);
    __syncthreads();
#pragma unroll
    for (int ks = 0; ks < 2; ++ks)
      mma_step(lt.A, lt.B, m0, ks * 32, lane, acc);
  }

  const int hi = lane >> 4, nl = lane & 15;
#pragma unroll
  for (int nt = 0; nt < 8; ++nt) {
    int n_g = nt * 16 + nl;
    float bia = bout[n_g];
#pragma unroll
    for (int r = 0; r < 8; ++r) {
      int row = r0 + m0 + r + 8 * hi;
      if (row < Nn) out[(size_t)row * OUTD + n_g] = acc[nt][r] + bia;
    }
  }
}

// ---------------------------------------------------------------------------
extern "C" void kernel_launch(void* const* d_in, const int* in_sizes, int n_in,
                              void* d_out, int out_size, void* d_ws, size_t ws_size,
                              hipStream_t stream) {
  const float* x      = (const float*)d_in[0];   // [N,128]
  const float* tgt    = (const float*)d_in[1];   // [E,128]
  const float* edgef  = (const float*)d_in[2];   // [E,64]
  const int*   uidx   = (const int*)d_in[3];     // [N]
  const int*   srcidx = (const int*)d_in[4];     // [E]
  const float* We_w   = (const float*)d_in[5];   // [3,128,192]
  const float* We_b   = (const float*)d_in[6];   // [3,128] == bcat[384]
  const float* W1     = (const float*)d_in[7];   // [3,128,128]
  const float* bn_g   = (const float*)d_in[8];   // [3,128]
  const float* bn_b   = (const float*)d_in[9];   // [3,128]
  const float* W2     = (const float*)d_in[10];  // [3,128,128]
  const float* Wout   = (const float*)d_in[11];  // [128,384]
  const float* bout   = (const float*)d_in[12];  // [128]

  const int Nn = in_sizes[0] / DN;
  const int E  = in_sizes[4];

  // ---- workspace carve ----
  size_t off = 0;
  char* base = (char*)d_ws;
  auto carve = [&](size_t bytes) -> char* {
    char* p = base + off;
    off += (bytes + 255) & ~(size_t)255;
    return p;
  };
  float*    agg    = (float*)carve((size_t)Nn * NCAT * 4);
  float*    conc   = (float*)carve((size_t)Nn * NCAT * 4);
  float*    zbuf   = (float*)carve((size_t)Nn * HIDD * 4);
  float*    bnsum  = (float*)carve(LNUM * HIDD * 4);
  float*    bnsq   = (float*)carve(LNUM * HIDD * 4);
  float*    bnA    = (float*)carve(LNUM * HIDD * 4);
  float*    bnB    = (float*)carve(LNUM * HIDD * 4);
  _Float16* WcatT  = (_Float16*)carve((size_t)NCAT * KTF * 2);
  _Float16* W1t    = (_Float16*)carve((size_t)LNUM * HIDD * HIDD * 2);
  _Float16* W2t    = (_Float16*)carve((size_t)LNUM * HIDD * HIDD * 2);
  _Float16* WoutT  = (_Float16*)carve((size_t)OUTD * NCAT * 2);

  // ---- prep: zero accumulators, convert weights to f16 ([n][k] order is the
  //      natural memory order of every weight tensor => elementwise convert) ----
  k_zero<<<2048, 256, 0, stream>>>(agg, (size_t)Nn * NCAT);
  k_zero<<<1, 256, 0, stream>>>(bnsum, LNUM * HIDD);
  k_zero<<<1, 256, 0, stream>>>(bnsq,  LNUM * HIDD);
  k_f2h<<<144, 256, 0, stream>>>(We_w, WcatT, NCAT * KTF);
  k_f2h<<<96,  256, 0, stream>>>(W1,   W1t,   LNUM * HIDD * HIDD);
  k_f2h<<<96,  256, 0, stream>>>(W2,   W2t,   LNUM * HIDD * HIDD);
  k_f2h<<<96,  256, 0, stream>>>(Wout, WoutT, OUTD * NCAT);

  // ---- edge GEMM + scatter (all 3 layers' aggregations in one pass) ----
  dim3 egrid((E + TILE_M - 1) / TILE_M, LNUM);
  k_edge_gemm_scatter<<<egrid, 256, 0, stream>>>(tgt, edgef, WcatT, We_b, srcidx, agg, E);

  // ---- node pipeline ----
  const int MB = (Nn + TILE_M - 1) / TILE_M;
  for (int l = 0; l < LNUM; ++l) {
    const float* hprev = (l == 0) ? x : (conc + (size_t)(l - 1) * HIDD);
    const int    ldh   = (l == 0) ? DN : NCAT;
    k_node_gemm_bnstat<<<MB, 256, 0, stream>>>(
        hprev, ldh, agg + (size_t)l * HIDD, W1t + (size_t)l * HIDD * HIDD,
        zbuf, bnsum + l * HIDD, bnsq + l * HIDD, Nn);
    k_bn_finalize<<<1, 128, 0, stream>>>(
        bnsum + l * HIDD, bnsq + l * HIDD, bn_g + l * HIDD, bn_b + l * HIDD,
        bnA + l * HIDD, bnB + l * HIDD, Nn);
    k_node_gemm_out<<<MB, 256, 0, stream>>>(
        zbuf, bnA + l * HIDD, bnB + l * HIDD, W2t + (size_t)l * HIDD * HIDD,
        conc + (size_t)l * HIDD, Nn);
  }

  // ---- readout GEMM ----
  float* outF = (float*)d_out;
  k_final_gemm<<<MB, 256, 0, stream>>>(conc, WoutT, bout, outF, Nn);

  // ---- second tuple output: passthrough indices ----
  int tail = out_size - Nn * OUTD;
  int cb = (Nn + 255) / 256;
  if (tail >= 2 * Nn) {
    k_widen_i64<<<cb, 256, 0, stream>>>(uidx, (long long*)(outF + (size_t)Nn * OUTD), Nn);
  } else if (tail >= Nn) {
    k_copy_i32<<<cb, 256, 0, stream>>>(uidx, (int*)(outF + (size_t)Nn * OUTD), Nn);
  }
}
